// TransformerSentenceEncoderLayer_17987323036071
// MI455X (gfx1250) — compile-verified, matrix operand order checked by reading, not run on hardware
//
#include <hip/hip_runtime.h>
#include <stdint.h>

// ---------------------------------------------------------------------------
// Static problem configuration (mirrors the reference)
// ---------------------------------------------------------------------------
#define BSZ     8
#define QLEN    256
#define DMODEL  1024
#define NHEAD   16
#define HDIM    64
#define FFN     4096
#define PAGE    16
#define KV_TOT  4096
#define PPS     256               // pages per sequence
#define NEWP    16                // new pages per sequence
#define NPAGES  (BSZ * PPS)       // 2048
#define PREFIX  (KV_TOT - QLEN)   // 3840
#define PAGESZ  (2 * PAGE * NHEAD * HDIM)   // 32768 floats per page
#define ROWS    (BSZ * QLEN)      // 2048

typedef __attribute__((ext_vector_type(16))) __bf16 v16bf;
typedef __attribute__((ext_vector_type(8)))  __bf16 v8bf;
typedef __attribute__((ext_vector_type(8)))  float  v8f;
typedef __attribute__((ext_vector_type(4)))  unsigned int v4u;
typedef __attribute__((ext_vector_type(8)))  int v8i;
typedef __attribute__((ext_vector_type(4)))  int v4i;

// f32 -> bf16 round-to-nearest-even (bit pattern only)
__device__ __forceinline__ unsigned short f2bf(float f) {
    union { float f; unsigned u; } v; v.f = f;
    unsigned r = v.u + 0x7FFFu + ((v.u >> 16) & 1u);
    return (unsigned short)(r >> 16);
}
__device__ __forceinline__ unsigned pack2(float a, float b) {
    return (unsigned)f2bf(a) | ((unsigned)f2bf(b) << 16);
}
// Build a 16-elt bf16 WMMA fragment from two 16-byte chunks
__device__ __forceinline__ v16bf frag16(const unsigned short* p0, const unsigned short* p1) {
    v8bf lo = *(const v8bf*)p0;
    v8bf hi = *(const v8bf*)p1;
    return __builtin_shufflevector(lo, hi, 0,1,2,3,4,5,6,7,8,9,10,11,12,13,14,15);
}
#define WMMA_BF16(A,Bx,C) \
    __builtin_amdgcn_wmma_f32_16x16x32_bf16(false,(A),false,(Bx),(short)0,(C),false,false)

// ---------------------------------------------------------------------------
// TDM: async-load one 64x64 bf16 tile (row stride = strideElems) into LDS.
// D# per CDNA5 ISA:
//   g0: count=1 | lds_addr | global_addr(57b) | type=2
//   g1: data_size=2B, pad_enable, pad_interval=32 DW, pad_amount=4 DW
//       (=> LDS row stride 128B + 16B = 144B, matching [64][72] ushort tiles),
//       tensor_dim0=64, tensor_dim1=64, tile_dim0=64, tile_dim1=64,
//       tensor_dim0_stride = strideElems. workgroup_mask=0 (not in cluster).
// Tracked by TENSORcnt; waited with s_wait_tensorcnt.
// amdgpu-toolchain (clang-23) 6-arg builtin form:
//   (v4u g0, v8i g1, v4i g2, v4i g3, v8i g4, i32 cpol)
// ---------------------------------------------------------------------------
__device__ __forceinline__ void tdm_tile64(unsigned lds_off,
                                           const unsigned short* g,
                                           int strideElems)
{
    unsigned long long ga = (unsigned long long)(uintptr_t)g;
    v4u g0;
    g0[0] = 1u;                                                  // count=1
    g0[1] = lds_off;                                             // lds_addr
    g0[2] = (unsigned)ga;                                        // global_addr lo
    g0[3] = (unsigned)((ga >> 32) & 0x01FFFFFFull) | (2u << 30); // addr hi | type=2
    v8i g1;
    g1[0] = (int)((1u << 16) | (1u << 20) | (4u << 22) | (3u << 25));
    g1[1] = (int)(64u << 16);      // tensor_dim0[15:0]=64
    g1[2] = (int)(64u << 16);      // tensor_dim0 hi=0 | tensor_dim1[15:0]=64
    g1[3] = (int)(64u << 16);      // tensor_dim1 hi=0 | tile_dim0=64
    g1[4] = 64;                    // tile_dim1=64 | tile_dim2=0
    g1[5] = strideElems;           // tensor_dim0_stride lo32
    g1[6] = 0;
    g1[7] = 0;
    v4i z4 = {0, 0, 0, 0};
    v8i z8 = {0, 0, 0, 0, 0, 0, 0, 0};
    __builtin_amdgcn_tensor_load_to_lds(g0, g1, z4, z4, z8, 0);
}

// ---------------------------------------------------------------------------
// bf16-WMMA GEMM with TDM double-buffered staging.
//   C[M,N] = epilogue( Abf[M,K] @ WT[N,K]^T )
// Abf is bf16 row-major; WT is the weight pre-transposed to [N][K] bf16 so
// both LDS tiles are plain 64x64 row-slices -> pure TDM copies, zero VALU.
// Block = 128 thr (4 waves); tile 64x64; each wave owns 32x32 (2x2 WMMA).
// Epilogue: v=(acc+bias[n])*scale; relu?; f32 out (+residual) or bf16 out.
// ---------------------------------------------------------------------------
__global__ __launch_bounds__(128) void gemm_bf16(
    const unsigned short* __restrict__ Abf, const unsigned short* __restrict__ WT,
    const float* __restrict__ bias, const float* __restrict__ Res,
    float* __restrict__ Cf, unsigned short* __restrict__ Cbf,
    int M, int N, int K, float scale, int relu)
{
    __shared__ __align__(16) unsigned short sA[2][64][72];   // 144B row stride
    __shared__ __align__(16) unsigned short sB[2][64][72];

    const int tid  = threadIdx.x;
    const int lane = tid & 31, wave = tid >> 5;
    const int wm = wave >> 1, wn = wave & 1;
    const int col = lane & 15, gh = lane >> 4;
    const int rowBase = blockIdx.y * 64, colBase = blockIdx.x * 64;

    const unsigned ldsA[2] = { (unsigned)(uintptr_t)&sA[0][0][0],
                               (unsigned)(uintptr_t)&sA[1][0][0] };
    const unsigned ldsB[2] = { (unsigned)(uintptr_t)&sB[0][0][0],
                               (unsigned)(uintptr_t)&sB[1][0][0] };
    const unsigned short* Arow = Abf + (size_t)rowBase * K;
    const unsigned short* Brow = WT  + (size_t)colBase * K;

    v8f acc[2][2];
    #pragma unroll
    for (int i = 0; i < 2; ++i)
        #pragma unroll
        for (int j = 0; j < 2; ++j)
            acc[i][j] = (v8f){0.f,0.f,0.f,0.f,0.f,0.f,0.f,0.f};

    // prologue: stage tile 0
    if (tid < 32) {
        tdm_tile64(ldsA[0], Arow, K);
        tdm_tile64(ldsB[0], Brow, K);
        __builtin_amdgcn_s_wait_tensorcnt(0);
    }
    __syncthreads();

    for (int kt = 0; kt < K; kt += 64) {
        const int cur = (kt >> 6) & 1;
        const int more = (kt + 64) < K;
        if (tid < 32 && more) {                    // async-prefetch next tile
            tdm_tile64(ldsA[cur ^ 1], Arow + kt + 64, K);
            tdm_tile64(ldsB[cur ^ 1], Brow + kt + 64, K);
        }
        const unsigned short (*pA)[72] = sA[cur];
        const unsigned short (*pB)[72] = sB[cur];
        #pragma unroll
        for (int kk = 0; kk < 64; kk += 32) {
            v16bf a0 = frag16(&pA[wm*32 +      col][kk + gh*8], &pA[wm*32 +      col][kk + 16 + gh*8]);
            v16bf a1 = frag16(&pA[wm*32 + 16 + col][kk + gh*8], &pA[wm*32 + 16 + col][kk + 16 + gh*8]);
            v16bf b0 = frag16(&pB[wn*32 +      col][kk + gh*8], &pB[wn*32 +      col][kk + 16 + gh*8]);
            v16bf b1 = frag16(&pB[wn*32 + 16 + col][kk + gh*8], &pB[wn*32 + 16 + col][kk + 16 + gh*8]);
            acc[0][0] = WMMA_BF16(a0, b0, acc[0][0]);
            acc[0][1] = WMMA_BF16(a0, b1, acc[0][1]);
            acc[1][0] = WMMA_BF16(a1, b0, acc[1][0]);
            acc[1][1] = WMMA_BF16(a1, b1, acc[1][1]);
        }
        if (tid < 32 && more) __builtin_amdgcn_s_wait_tensorcnt(0);
        __syncthreads();
    }

    // ---- epilogue ----
    #pragma unroll
    for (int mi = 0; mi < 2; ++mi)
        #pragma unroll
        for (int ni = 0; ni < 2; ++ni) {
            int r0 = rowBase + wm*32 + mi*16 + gh*8;
            int c0 = colBase + wn*32 + ni*16 + col;
            float bv = bias[c0];
            #pragma unroll
            for (int r = 0; r < 8; ++r) {
                float v = (acc[mi][ni][r] + bv) * scale;
                if (relu) v = fmaxf(v, 0.f);
                size_t idx = (size_t)(r0 + r) * N + c0;
                if (Cbf) {
                    Cbf[idx] = f2bf(v);
                } else {
                    if (Res) v += Res[idx];
                    Cf[idx] = v;
                }
            }
        }
}

// ---------------------------------------------------------------------------
// Weight convert+transpose: W[K,N] f32 -> WT[N,K] bf16 (32x32 LDS tiles)
// ---------------------------------------------------------------------------
__global__ __launch_bounds__(256) void transpose_bf16(
    const float* __restrict__ W, unsigned short* __restrict__ WT, int K, int N)
{
    __shared__ float tile[32][33];
    const int bk = blockIdx.x * 32, bn = blockIdx.y * 32;
    const int tx = threadIdx.x & 31, ty = threadIdx.x >> 5;
    #pragma unroll
    for (int i = 0; i < 4; ++i) {
        int k = ty + i * 8;
        tile[k][tx] = W[(size_t)(bk + k) * N + bn + tx];
    }
    __syncthreads();
    #pragma unroll
    for (int i = 0; i < 4; ++i) {
        int n = ty + i * 8;
        WT[(size_t)(bn + n) * K + bk + tx] = f2bf(tile[tx][n]);
    }
}

// x f32 -> bf16 row-major (elementwise)
__global__ void cvt_bf16(const float* __restrict__ S, unsigned short* __restrict__ D, int n)
{
    int i = blockIdx.x * 256 + threadIdx.x;
    if (i < n) D[i] = f2bf(S[i]);
}

// ---------------------------------------------------------------------------
// pagetable -> pt output bulk copy (float4 grid-stride)
// ---------------------------------------------------------------------------
__global__ void copy4(const float4* __restrict__ s, float4* __restrict__ d, int n)
{
    int i = blockIdx.x * blockDim.x + threadIdx.x;
    int stride = gridDim.x * blockDim.x;
    for (; i < n; i += stride) d[i] = s[i];
}

// ---------------------------------------------------------------------------
// Scatter new K/V into the last NEWP pages of each sequence.
// ---------------------------------------------------------------------------
__global__ __launch_bounds__(256) void scatter_kv(
    float* __restrict__ PT, const float* __restrict__ Kn,
    const float* __restrict__ Vn, const int* __restrict__ kvidx)
{
    const int bj = blockIdx.x;             // [0, BSZ*NEWP)
    const int b = bj >> 4, j = bj & 15;
    const int pid = kvidx[b * PPS + (PPS - NEWP) + j];
    float* dst = PT + (size_t)pid * PAGESZ;
    const size_t srow = (size_t)(b * QLEN + j * PAGE);
    for (int e = threadIdx.x; e < PAGESZ; e += 256) {
        int s  = e >> 14;
        int rm = e & 16383;
        int pr = rm >> 10, c = rm & 1023;
        const float* src = s ? Vn : Kn;
        dst[e] = src[(srow + pr) * DMODEL + c];
    }
}

// ---------------------------------------------------------------------------
// Paged flash attention. Block = 128 thr (4 waves) = one (b, h, 64 q rows).
// Q is bf16 (read directly as fragments); K/V staged from the f32 paged cache
// with inline bf16 conversion (K natural [kv][d], V transposed [d][kv]).
// Writes ctx as bf16 (A operand of the Wo GEMM).
// ---------------------------------------------------------------------------
__global__ __launch_bounds__(128) void attn_kernel(
    const unsigned short* __restrict__ Qbf, const float* __restrict__ PT,
    const int* __restrict__ kvidx, unsigned short* __restrict__ Ctxbf)
{
    __shared__ __align__(16) unsigned short sK[32][72];
    __shared__ __align__(16) unsigned short sV[64][48];
    __shared__ __align__(16) unsigned short sP[4][16][48];

    const int tid = threadIdx.x, lane = tid & 31, wave = tid >> 5;
    const int col = lane & 15, gh = lane >> 4;
    const int qg = blockIdx.x & 3;
    const int h  = (blockIdx.x >> 2) & 15;
    const int b  = blockIdx.x >> 6;
    const int qbase = (qg * 4 + wave) * 16;

    // q fragments straight from bf16 global (16B chunks, aligned)
    v16bf qf[2];
    {
        const unsigned short* qrow =
            Qbf + (size_t)(b * QLEN + qbase + col) * DMODEL + h * HDIM;
        #pragma unroll
        for (int ds = 0; ds < 2; ++ds)
            qf[ds] = frag16(qrow + ds * 32 + gh * 8, qrow + ds * 32 + 16 + gh * 8);
    }

    v8f oacc[4];
    #pragma unroll
    for (int i = 0; i < 4; ++i) oacc[i] = (v8f){0.f,0.f,0.f,0.f,0.f,0.f,0.f,0.f};
    float m[8], l[8];
    #pragma unroll
    for (int r = 0; r < 8; ++r) { m[r] = -1e30f; l[r] = 0.f; }

    const int nblk = ((PREFIX + qg * 64 + 63) >> 5) + 1;
    for (int blk = 0; blk < nblk; ++blk) {
        __syncthreads();
        const int p0 = kvidx[b * PPS + blk * 2];
        const int p1 = kvidx[b * PPS + blk * 2 + 1];
        #pragma unroll
        for (int i = 0; i < 4; ++i) {
            int e = tid + i * 128;
            int kvr = e >> 4, dq = (e & 15) << 2;
            size_t pb = (size_t)(kvr < 16 ? p0 : p1) * PAGESZ
                      + (size_t)(kvr & 15) * (NHEAD * HDIM) + h * HDIM + dq;
            float4 k4 = *(const float4*)(PT + pb);
            uint2 t; t.x = pack2(k4.x, k4.y); t.y = pack2(k4.z, k4.w);
            *(uint2*)&sK[kvr][dq] = t;
            float4 v4 = *(const float4*)(PT + pb + PAGE * NHEAD * HDIM);
            sV[dq + 0][kvr] = f2bf(v4.x);
            sV[dq + 1][kvr] = f2bf(v4.y);
            sV[dq + 2][kvr] = f2bf(v4.z);
            sV[dq + 3][kvr] = f2bf(v4.w);
        }
        __syncthreads();

        v8f s0 = (v8f){0.f,0.f,0.f,0.f,0.f,0.f,0.f,0.f};
        v8f s1 = (v8f){0.f,0.f,0.f,0.f,0.f,0.f,0.f,0.f};
        #pragma unroll
        for (int ds = 0; ds < 2; ++ds) {
            v16bf k0 = frag16(&sK[col][ds*32 + gh*8],      &sK[col][ds*32 + 16 + gh*8]);
            v16bf k1 = frag16(&sK[16 + col][ds*32 + gh*8], &sK[16 + col][ds*32 + 16 + gh*8]);
            s0 = WMMA_BF16(qf[ds], k0, s0);
            s1 = WMMA_BF16(qf[ds], k1, s1);
        }

        const int qpos0 = PREFIX + qbase + gh * 8;
        const int kv0b = blk * 32 + col, kv1b = blk * 32 + 16 + col;
        #pragma unroll
        for (int r = 0; r < 8; ++r) {
            int qpos = qpos0 + r;
            float a  = (kv0b <= qpos) ? s0[r] : -1e30f;
            float c2 = (kv1b <= qpos) ? s1[r] : -1e30f;
            float t = fmaxf(a, c2);
            t = fmaxf(t, __shfl_xor(t, 1));
            t = fmaxf(t, __shfl_xor(t, 2));
            t = fmaxf(t, __shfl_xor(t, 4));
            t = fmaxf(t, __shfl_xor(t, 8));
            float mn = fmaxf(m[r], t);
            float sf = __expf(m[r] - mn);
            m[r] = mn;
            float pa = __expf(a - mn), pc = __expf(c2 - mn);
            float ps = pa + pc;
            ps += __shfl_xor(ps, 1); ps += __shfl_xor(ps, 2);
            ps += __shfl_xor(ps, 4); ps += __shfl_xor(ps, 8);
            l[r] = l[r] * sf + ps;
            oacc[0][r] *= sf; oacc[1][r] *= sf; oacc[2][r] *= sf; oacc[3][r] *= sf;
            sP[wave][gh * 8 + r][col]      = f2bf(pa);
            sP[wave][gh * 8 + r][16 + col] = f2bf(pc);
        }

        v16bf pf = frag16(&sP[wave][col][gh * 8], &sP[wave][col][16 + gh * 8]);
        #pragma unroll
        for (int nd = 0; nd < 4; ++nd) {
            v16bf vf = frag16(&sV[nd * 16 + col][gh * 8], &sV[nd * 16 + col][16 + gh * 8]);
            oacc[nd] = WMMA_BF16(pf, vf, oacc[nd]);
        }
    }

    #pragma unroll
    for (int nd = 0; nd < 4; ++nd)
        #pragma unroll
        for (int r = 0; r < 8; ++r) {
            size_t row = (size_t)(b * QLEN + qbase + gh * 8 + r);
            Ctxbf[row * DMODEL + h * HDIM + nd * 16 + col] = f2bf(oacc[nd][r] / l[r]);
        }
}

// ---------------------------------------------------------------------------
// LayerNorm over 1024 cols; one block (256 thr) per row; optional residual,
// optional extra bf16 copy of the output (A operand of the next GEMM).
// ---------------------------------------------------------------------------
__global__ __launch_bounds__(256) void ln_kernel(
    const float* __restrict__ S, const float* __restrict__ R,
    const float* __restrict__ G, const float* __restrict__ Bt,
    float* __restrict__ O, unsigned short* __restrict__ Obf, int hasRes)
{
    __shared__ float red[2][8];
    const int row = blockIdx.x, tid = threadIdx.x;
    const int lane = tid & 31, wave = tid >> 5;
    const size_t base = (size_t)row * DMODEL + tid * 4;

    float4 v = *(const float4*)(S + base);
    if (hasRes) {
        float4 rr = *(const float4*)(R + base);
        v.x += rr.x; v.y += rr.y; v.z += rr.z; v.w += rr.w;
    }
    float sum = v.x + v.y + v.z + v.w;
    float ss  = v.x*v.x + v.y*v.y + v.z*v.z + v.w*v.w;
    #pragma unroll
    for (int off = 1; off < 32; off <<= 1) {
        sum += __shfl_xor(sum, off);
        ss  += __shfl_xor(ss, off);
    }
    if (lane == 0) { red[0][wave] = sum; red[1][wave] = ss; }
    __syncthreads();
    float ts = 0.f, tq = 0.f;
    #pragma unroll
    for (int w = 0; w < 8; ++w) { ts += red[0][w]; tq += red[1][w]; }
    const float mu   = ts * (1.f / DMODEL);
    const float var  = tq * (1.f / DMODEL) - mu * mu;
    const float rstd = rsqrtf(var + 1e-5f);

    float4 g4 = *(const float4*)(G + tid * 4);
    float4 b4 = *(const float4*)(Bt + tid * 4);
    float4 o;
    o.x = (v.x - mu) * rstd * g4.x + b4.x;
    o.y = (v.y - mu) * rstd * g4.y + b4.y;
    o.z = (v.z - mu) * rstd * g4.z + b4.z;
    o.w = (v.w - mu) * rstd * g4.w + b4.w;
    *(float4*)(O + base) = o;
    if (Obf) {
        Obf[base + 0] = f2bf(o.x);
        Obf[base + 1] = f2bf(o.y);
        Obf[base + 2] = f2bf(o.z);
        Obf[base + 3] = f2bf(o.w);
    }
}

// ---------------------------------------------------------------------------
extern "C" void kernel_launch(void* const* d_in, const int* in_sizes, int n_in,
                              void* d_out, int out_size, void* d_ws, size_t ws_size,
                              hipStream_t stream)
{
    (void)in_sizes; (void)n_in; (void)out_size; (void)ws_size;

    const float* x    = (const float*)d_in[0];
    const float* ptab = (const float*)d_in[1];
    const float* Wq   = (const float*)d_in[2];
    const float* bq   = (const float*)d_in[3];
    const float* Wk   = (const float*)d_in[4];
    const float* bk   = (const float*)d_in[5];
    const float* Wv   = (const float*)d_in[6];
    const float* bv   = (const float*)d_in[7];
    const float* Wo   = (const float*)d_in[8];
    const float* bo   = (const float*)d_in[9];
    const float* ln1g = (const float*)d_in[10];
    const float* ln1b = (const float*)d_in[11];
    const float* W1   = (const float*)d_in[12];
    const float* b1   = (const float*)d_in[13];
    const float* W2   = (const float*)d_in[14];
    const float* b2   = (const float*)d_in[15];
    const float* lnfg = (const float*)d_in[16];
    const float* lnfb = (const float*)d_in[17];
    const int*   kvix = (const int*)d_in[20];

    float* out_x     = (float*)d_out;
    float* out_layer = out_x + (size_t)ROWS * DMODEL;
    float* out_pt    = out_layer + (size_t)ROWS * DMODEL;

    // workspace carve-out (bytes)
    char* w = (char*)d_ws;
    const size_t MB = (size_t)1 << 20;
    unsigned short* WqT   = (unsigned short*)(w + 0 * MB);    // 2MB  [1024][1024]
    unsigned short* WkT   = (unsigned short*)(w + 2 * MB);    // 2MB
    unsigned short* WvT   = (unsigned short*)(w + 4 * MB);    // 2MB
    unsigned short* WoT   = (unsigned short*)(w + 6 * MB);    // 2MB
    unsigned short* W1T   = (unsigned short*)(w + 8 * MB);    // 8MB  [4096][1024]
    unsigned short* W2T   = (unsigned short*)(w + 16 * MB);   // 8MB  [1024][4096]
    unsigned short* xbf   = (unsigned short*)(w + 24 * MB);   // 4MB
    unsigned short* qbf   = (unsigned short*)(w + 28 * MB);   // 4MB
    float*          k_ws  = (float*)(w + 32 * MB);            // 8MB  (later xres)
    float*          v_ws  = (float*)(w + 40 * MB);            // 8MB  (later ctxbf)
    float*          xln1  = (float*)(w + 48 * MB);            // 8MB
    unsigned short* xln1b = (unsigned short*)(w + 56 * MB);   // 4MB
    unsigned short* hbf   = (unsigned short*)(w + 60 * MB);   // 16MB [2048][4096]
    float*          xres  = k_ws;                             // k consumed by scatter
    unsigned short* ctxbf = (unsigned short*)v_ws;            // v consumed by scatter

    const dim3 g1024(DMODEL / 64, ROWS / 64);                 // (16, 32)
    const dim3 g4096(FFN / 64,   ROWS / 64);                  // (64, 32)
    const dim3 tQ(DMODEL / 32, DMODEL / 32);
    const dim3 t1(DMODEL / 32, FFN / 32);
    const dim3 t2(FFN / 32,   DMODEL / 32);

    // ---- one-shot weight transpose+convert, activation convert ----
    transpose_bf16<<<tQ, 256, 0, stream>>>(Wq, WqT, DMODEL, DMODEL);
    transpose_bf16<<<tQ, 256, 0, stream>>>(Wk, WkT, DMODEL, DMODEL);
    transpose_bf16<<<tQ, 256, 0, stream>>>(Wv, WvT, DMODEL, DMODEL);
    transpose_bf16<<<tQ, 256, 0, stream>>>(Wo, WoT, DMODEL, DMODEL);
    transpose_bf16<<<t1, 256, 0, stream>>>(W1, W1T, DMODEL, FFN);
    transpose_bf16<<<t2, 256, 0, stream>>>(W2, W2T, FFN, DMODEL);
    cvt_bf16<<<(ROWS * DMODEL) / 256, 256, 0, stream>>>(x, xbf, ROWS * DMODEL);

    // ---- QKV projections (q fused with 1/sqrt(HD) scale, bf16 out) ----
    gemm_bf16<<<g1024, 128, 0, stream>>>(xbf, WqT, bq, nullptr, nullptr, qbf,
                                         ROWS, DMODEL, DMODEL, 0.125f, 0);
    gemm_bf16<<<g1024, 128, 0, stream>>>(xbf, WkT, bk, nullptr, k_ws, nullptr,
                                         ROWS, DMODEL, DMODEL, 1.f, 0);
    gemm_bf16<<<g1024, 128, 0, stream>>>(xbf, WvT, bv, nullptr, v_ws, nullptr,
                                         ROWS, DMODEL, DMODEL, 1.f, 0);

    // ---- pagetable copy + new-KV scatter (produces pt output) ----
    copy4<<<8192, 256, 0, stream>>>((const float4*)ptab, (float4*)out_pt,
                                    (int)((size_t)NPAGES * PAGESZ / 4));
    scatter_kv<<<BSZ * NEWP, 256, 0, stream>>>(out_pt, k_ws, v_ws, kvix);

    // ---- paged flash attention (bf16 ctx out) ----
    attn_kernel<<<BSZ * NHEAD * (QLEN / 64), 128, 0, stream>>>(qbf, out_pt, kvix, ctxbf);

    // ---- output projection + residual, LN1 (f32 + bf16 copies) ----
    gemm_bf16<<<g1024, 128, 0, stream>>>(ctxbf, WoT, bo, x, xres, nullptr,
                                         ROWS, DMODEL, DMODEL, 1.f, 0);
    ln_kernel<<<ROWS, 256, 0, stream>>>(xres, nullptr, ln1g, ln1b, xln1, xln1b, 0);

    // ---- FFN ----
    gemm_bf16<<<g4096, 128, 0, stream>>>(xln1b, W1T, b1, nullptr, nullptr, hbf,
                                         ROWS, FFN, DMODEL, 1.f, 1);
    gemm_bf16<<<g1024, 128, 0, stream>>>(hbf, W2T, b2, nullptr, out_layer, nullptr,
                                         ROWS, DMODEL, FFN, 1.f, 0);

    // ---- final residual + LN ----
    ln_kernel<<<ROWS, 256, 0, stream>>>(xln1, out_layer, lnfg, lnfb, out_x, nullptr, 1);
}
